// TransformerBlock_28621662060705
// MI455X (gfx1250) — matrix operand order repaired and checked by reading, not compiled, not run on hardware
//
#include <hip/hip_runtime.h>
#include <math.h>

// ---------------------------------------------------------------------------
// MI455X (gfx1250) transformer block: bf16 WMMA + async global->LDS staging.
// B=4 S=2048 D=1024 H=16 HD=64 DFF=4096, fp32 I/O, bf16 tensor-core compute.
// ---------------------------------------------------------------------------

typedef __bf16 bf16;
typedef float v8f  __attribute__((ext_vector_type(8)));
typedef bf16  v16bf __attribute__((ext_vector_type(16)));

struct U32x8 { uint4 lo, hi; };

static constexpr int Bc = 4, Sc = 2048, Dc = 1024, Hc = 16, HDc = 64, DFFc = 4096;
static constexpr int Mc = Bc * Sc; // 8192 rows

// ---- CDNA5 async global->LDS copy (ASYNCcnt) -------------------------------
__device__ inline void async_copy_b128(unsigned lds_addr, const bf16* gptr) {
  // per-lane: LDS[lds_addr] <- MEM[gptr], 16 bytes, tracked by ASYNCcnt
  asm volatile("global_load_async_to_lds_b128 %0, %1, off"
               :: "v"(lds_addr), "v"(gptr)
               : "memory");
}
__device__ inline void wait_async0() {
  asm volatile("s_wait_asynccnt 0x0" ::: "memory");
}
__device__ inline unsigned lds_addr_of(const void* p) {
  // shared aperture occupies the high 32 bits of the flat address; low 32
  // bits are the byte offset within the workgroup LDS allocation.
  return (unsigned)(uintptr_t)p;
}

// ---- WMMA wrapper ----------------------------------------------------------
__device__ inline v8f wmma_bf16(v16bf a, v16bf b, v8f c) {
  return __builtin_amdgcn_wmma_f32_16x16x32_bf16(
      /*neg_a=*/false, a, /*neg_b=*/false, b,
      /*c_mod=*/(short)0, c, /*reuse_a=*/false, /*reuse_b=*/false);
}

// A fragment (16x32 bf16): lane holds row M=ln; element e is K = 16g + e
// (16 contiguous bf16).
__device__ inline v16bf load_a_frag(const bf16* base, int row, int kbase,
                                    int g, int ldk) {
  const bf16* p = base + row * ldk + kbase;
  U32x8 t;
  t.lo = *reinterpret_cast<const uint4*>(p + 16 * g);
  t.hi = *reinterpret_cast<const uint4*>(p + 16 * g + 8);
  return __builtin_bit_cast(v16bf, t);
}

// B fragment (32x16 bf16) from a K-contiguous (transposed) tile:
// lane holds col N=ln; element e is K = 16g + e -> 16 contiguous bf16.
__device__ inline v16bf load_b_frag(const bf16* baseT, int n, int kbase,
                                    int g, int ldk) {
  const bf16* p = baseT + n * ldk + kbase + 16 * g;
  U32x8 t;
  t.lo = reinterpret_cast<const uint4*>(p)[0];
  t.hi = reinterpret_cast<const uint4*>(p)[1];
  return __builtin_bit_cast(v16bf, t);
}

__device__ inline float gelu_f(float x) {
  const float c = 0.79788456080286535588f; // sqrt(2/pi)
  return 0.5f * x * (1.0f + tanhf(c * (x + 0.044715f * x * x * x)));
}

__device__ inline float rowmax16(float v) {
  #pragma unroll
  for (int m = 1; m < 16; m <<= 1) v = fmaxf(v, __shfl_xor(v, m, 32));
  return v;
}
__device__ inline float rowsum16(float v) {
  #pragma unroll
  for (int m = 1; m < 16; m <<= 1) v += __shfl_xor(v, m, 32);
  return v;
}

// ---- fp32 -> bf16 convert --------------------------------------------------
__global__ void cvt_f32_bf16_kernel(const float* __restrict__ in,
                                    bf16* __restrict__ out, int n) {
  for (int i = blockIdx.x * blockDim.x + threadIdx.x; i < n;
       i += gridDim.x * blockDim.x)
    out[i] = (bf16)in[i];
}

// ---- LayerNorm (fp32 in, bf16 out), one block per row of D=1024 ------------
__global__ void __launch_bounds__(256)
ln_bf16_kernel(const float* __restrict__ x, const float* __restrict__ sc,
               const float* __restrict__ sh, bf16* __restrict__ out) {
  const int row = blockIdx.x, tid = threadIdx.x;
  const float* xr = x + (size_t)row * Dc;
  float vals[4], s = 0.f, q = 0.f;
  #pragma unroll
  for (int i = 0; i < 4; ++i) {
    float v = xr[tid + 256 * i];
    vals[i] = v; s += v; q += v * v;
  }
  #pragma unroll
  for (int m = 1; m < 32; m <<= 1) {
    s += __shfl_xor(s, m, 32);
    q += __shfl_xor(q, m, 32);
  }
  __shared__ float ss[8], sq[8];
  const int w = tid >> 5, lane = tid & 31;
  if (lane == 0) { ss[w] = s; sq[w] = q; }
  __syncthreads();
  if (tid == 0) {
    float S = 0.f, Q = 0.f;
    for (int i = 0; i < 8; ++i) { S += ss[i]; Q += sq[i]; }
    ss[0] = S; sq[0] = Q;
  }
  __syncthreads();
  const float mean = ss[0] * (1.0f / Dc);
  const float var  = sq[0] * (1.0f / Dc) - mean * mean;
  const float rstd = rsqrtf(var + 1e-5f);
  #pragma unroll
  for (int i = 0; i < 4; ++i) {
    int c = tid + 256 * i;
    out[(size_t)row * Dc + c] = (bf16)(sc[c] * (vals[i] - mean) * rstd + sh[c]);
  }
}

// ---- bf16 WMMA GEMM: C[M,N] = A[M,K]*B[K,N] (+bias)(+gelu)(+resid) ---------
// 256 threads = 8 waves; block tile 128x128; wave tile 32x64 (2x4 wmma).
// Double-buffered LDS; A tile staged with async global->LDS; one barrier/step.
template <bool BIAS, bool GELU, bool RESID, bool OUTF32>
__global__ void __launch_bounds__(256)
gemm_wmma_kernel(const bf16* __restrict__ A, const bf16* __restrict__ Bm,
                 const float* __restrict__ bias, const float* __restrict__ resid,
                 void* __restrict__ outp, int M, int N, int K) {
  constexpr int BM = 128, BN = 128, BK = 32;
  __shared__ alignas(16) bf16 As[2][BM * BK];   // [m][k]
  __shared__ alignas(16) bf16 Bt[2][BN * BK];   // [n][k] (transposed)
  const int tid = threadIdx.x;
  const int lane = tid & 31, w = tid >> 5;
  const int g = lane >> 4, ln = lane & 15;
  const int wm = w & 3, wn = w >> 2;
  const int m0 = blockIdx.y * BM, n0 = blockIdx.x * BN;

  v8f acc[2][4];
  #pragma unroll
  for (int i = 0; i < 2; ++i)
    #pragma unroll
    for (int j = 0; j < 4; ++j) acc[i][j] = v8f{};

  // A staging: 2 threads/row, 32 B (16 bf16) each = 2 async b128 per thread
  const int arow = tid >> 1, ahalf = tid & 1;
  const bf16* agbase = A + (size_t)(m0 + arow) * K + ahalf * 16;
  const unsigned alds[2] = {
      lds_addr_of(&As[0][arow * BK + ahalf * 16]),
      lds_addr_of(&As[1][arow * BK + ahalf * 16])};

  // B staging: thread covers row kk = tid&31, 16 cols at nb = (tid>>5)*16
  const int bkk = tid & 31, bnb = (tid >> 5) * 16;
  const bf16* bgbase = Bm + (size_t)bkk * N + n0 + bnb;

  union BRegs { uint4 u[2]; bf16 h[16]; };

  const int KT = K / BK;

  // prologue: tile 0 in flight
  async_copy_b128(alds[0], agbase);
  async_copy_b128(alds[0] + 16, agbase + 8);
  BRegs br;
  {
    const uint4* p = reinterpret_cast<const uint4*>(bgbase);
    br.u[0] = p[0]; br.u[1] = p[1];
  }

  for (int kt = 0; kt < KT; ++kt) {
    const int cur = kt & 1;
    // transpose-store prefetched B regs into Bt[cur]
    #pragma unroll
    for (int e = 0; e < 16; ++e) Bt[cur][(bnb + e) * BK + bkk] = br.h[e];
    wait_async0();        // As[cur] landed (this wave's async ops)
    __syncthreads();      // all staging visible block-wide
    if (kt + 1 < KT) {    // prefetch next tile, overlapped with WMMAs below
      const bf16* asrc = agbase + (kt + 1) * BK;
      async_copy_b128(alds[cur ^ 1], asrc);
      async_copy_b128(alds[cur ^ 1] + 16, asrc + 8);
      const uint4* p = reinterpret_cast<const uint4*>(
          bgbase + (size_t)(kt + 1) * BK * N);
      br.u[0] = p[0]; br.u[1] = p[1];
    }
    v16bf afr[2];
    #pragma unroll
    for (int tm = 0; tm < 2; ++tm)
      afr[tm] = load_a_frag(As[cur], wm * 32 + tm * 16 + ln, 0, g, BK);
    #pragma unroll
    for (int tn = 0; tn < 4; ++tn) {
      v16bf bfr = load_b_frag(Bt[cur], wn * 64 + tn * 16 + ln, 0, g, BK);
      #pragma unroll
      for (int tm = 0; tm < 2; ++tm)
        acc[tm][tn] = wmma_bf16(afr[tm], bfr, acc[tm][tn]);
    }
  }

  // epilogue: C layout row = r + 8*g, col = ln within each 16x16 tile
  #pragma unroll
  for (int tm = 0; tm < 2; ++tm)
    #pragma unroll
    for (int tn = 0; tn < 4; ++tn)
      #pragma unroll
      for (int r = 0; r < 8; ++r) {
        const int row = m0 + wm * 32 + tm * 16 + r + 8 * g;
        const int col = n0 + wn * 64 + tn * 16 + ln;
        float v = acc[tm][tn][r];
        if (BIAS)  v += bias[col];
        if (GELU)  v = gelu_f(v);
        if (RESID) v += resid[(size_t)row * N + col];
        if (OUTF32)
          reinterpret_cast<float*>(outp)[(size_t)row * N + col] = v;
        else
          reinterpret_cast<bf16*>(outp)[(size_t)row * N + col] = (bf16)v;
      }
}

// ---- causal flash attention, bf16 WMMA, HD=64 ------------------------------
// block = 128 threads (4 waves); 64-query tile per block; wave owns 16 rows.
// Q/K tiles staged via async global->LDS; V staged manually (transpose).
__global__ void __launch_bounds__(128)
flash_attn_kernel(const bf16* __restrict__ Q, const bf16* __restrict__ K,
                  const bf16* __restrict__ V, bf16* __restrict__ O) {
  __shared__ alignas(16) bf16 Qs[64 * 64];     // [q][d]
  __shared__ alignas(16) bf16 Kt[64 * 64];     // [key][d]  (B^T for scores)
  __shared__ alignas(16) bf16 Vt[64 * 64];     // [d][key]  (B^T for ctx)
  __shared__ alignas(16) bf16 Ps[4 * 16 * 64]; // per-wave P relayout [q][key]
  const int tid = threadIdx.x;
  const int lane = tid & 31, wid = tid >> 5;
  const int g = lane >> 4, ln = lane & 15;
  const int qt = blockIdx.x, h = blockIdx.y, b = blockIdx.z;
  const int q0 = qt * 64;
  const size_t bofs = ((size_t)b * Sc) * Dc + (size_t)h * HDc;

  const int crow = tid >> 1, chalf = tid & 1; // tile copy: 2 thr/row, 64 B each

  { // preload Q tile (async)
    const bf16* src = Q + bofs + (size_t)(q0 + crow) * Dc + chalf * 32;
    const unsigned dst = lds_addr_of(Qs + crow * 64 + chalf * 32);
    #pragma unroll
    for (int i = 0; i < 4; ++i) async_copy_b128(dst + 16 * i, src + 8 * i);
  }

  v8f o[4];
  #pragma unroll
  for (int i = 0; i < 4; ++i) o[i] = v8f{};
  float mrun[8], lrun[8];
  #pragma unroll
  for (int r = 0; r < 8; ++r) { mrun[r] = -1e30f; lrun[r] = 0.f; }

  for (int jt = 0; jt <= qt; ++jt) {
    const int k0 = jt * 64;
    __syncthreads(); // previous tile reads done before overwrite
    { // K tile -> Kt[key][d] (async)
      const bf16* src = K + bofs + (size_t)(k0 + crow) * Dc + chalf * 32;
      const unsigned dst = lds_addr_of(Kt + crow * 64 + chalf * 32);
      #pragma unroll
      for (int i = 0; i < 4; ++i) async_copy_b128(dst + 16 * i, src + 8 * i);
    }
    { // V tile transposed -> Vt[d][key] (manual: scatter)
      union { uint4 u[4]; bf16 hh[32]; } cv;
      const uint4* src = reinterpret_cast<const uint4*>(
          V + bofs + (size_t)(k0 + crow) * Dc + chalf * 32);
      cv.u[0] = src[0]; cv.u[1] = src[1]; cv.u[2] = src[2]; cv.u[3] = src[3];
      #pragma unroll
      for (int e = 0; e < 32; ++e) Vt[(chalf * 32 + e) * 64 + crow] = cv.hh[e];
    }
    wait_async0();    // this wave's Q/K async ops landed
    __syncthreads();  // staging visible block-wide

    // scores S = Q * K^T (16 q-rows x 64 keys per wave)
    v8f s[4];
    #pragma unroll
    for (int i = 0; i < 4; ++i) s[i] = v8f{};
    #pragma unroll
    for (int kk = 0; kk < 2; ++kk) {
      v16bf afr = load_a_frag(Qs + wid * 16 * 64, ln, kk * 32, g, 64);
      #pragma unroll
      for (int tn = 0; tn < 4; ++tn) {
        v16bf bfr = load_b_frag(Kt, tn * 16 + ln, kk * 32, g, 64);
        s[tn] = wmma_bf16(afr, bfr, s[tn]);
      }
    }
    // scale + causal mask (only on diagonal tile)
    #pragma unroll
    for (int tn = 0; tn < 4; ++tn)
      #pragma unroll
      for (int r = 0; r < 8; ++r) {
        float v = s[tn][r] * 0.125f; // 1/sqrt(64)
        if (jt == qt) {
          const int key = k0 + tn * 16 + ln;
          const int qr  = q0 + wid * 16 + r + 8 * g;
          if (key > qr) v = -1e30f;
        }
        s[tn][r] = v;
      }
    // online softmax per row (row spans 16 lanes of a half-wave)
    #pragma unroll
    for (int r = 0; r < 8; ++r) {
      float mx = fmaxf(fmaxf(s[0][r], s[1][r]), fmaxf(s[2][r], s[3][r]));
      mx = rowmax16(mx);
      const float mnew  = fmaxf(mrun[r], mx);
      const float alpha = expf(mrun[r] - mnew);
      float lsum = 0.f;
      #pragma unroll
      for (int tn = 0; tn < 4; ++tn) {
        const float p = expf(s[tn][r] - mnew);
        lsum += p;
        Ps[(wid * 16 + r + 8 * g) * 64 + tn * 16 + ln] = (bf16)p;
      }
      lsum = rowsum16(lsum);
      lrun[r] = lrun[r] * alpha + lsum;
      mrun[r] = mnew;
      #pragma unroll
      for (int dt = 0; dt < 4; ++dt) o[dt][r] *= alpha;
    }
    // ctx += P * V (per-wave Ps relayout -> A frags; in-wave DS ordering)
    #pragma unroll
    for (int kk = 0; kk < 2; ++kk) {
      v16bf afr = load_a_frag(Ps + wid * 16 * 64, ln, kk * 32, g, 64);
      #pragma unroll
      for (int dt = 0; dt < 4; ++dt) {
        v16bf bfr = load_b_frag(Vt, dt * 16 + ln, kk * 32, g, 64);
        o[dt] = wmma_bf16(afr, bfr, o[dt]);
      }
    }
  }

  #pragma unroll
  for (int r = 0; r < 8; ++r) {
    const float inv = 1.0f / lrun[r];
    const int qr = q0 + wid * 16 + r + 8 * g;
    #pragma unroll
    for (int dt = 0; dt < 4; ++dt)
      O[bofs + (size_t)qr * Dc + dt * 16 + ln] = (bf16)(o[dt][r] * inv);
  }
}

// ---------------------------------------------------------------------------
extern "C" void kernel_launch(void* const* d_in, const int* in_sizes, int n_in,
                              void* d_out, int out_size, void* d_ws,
                              size_t ws_size, hipStream_t stream) {
  (void)in_sizes; (void)n_in; (void)out_size; (void)ws_size;
  const float* x    = (const float*)d_in[0];
  const float* Wq   = (const float*)d_in[1];
  const float* Wk   = (const float*)d_in[2];
  const float* Wv   = (const float*)d_in[3];
  const float* Wo   = (const float*)d_in[4];
  const float* bo   = (const float*)d_in[5];
  const float* W1   = (const float*)d_in[6];
  const float* b1   = (const float*)d_in[7];
  const float* W2   = (const float*)d_in[8];
  const float* b2   = (const float*)d_in[9];
  const float* ln1s = (const float*)d_in[10];
  const float* ln1b = (const float*)d_in[11];
  const float* ln2s = (const float*)d_in[12];
  const float* ln2b = (const float*)d_in[13];

  const size_t MiB = 1024ull * 1024ull;
  char* ws = (char*)d_ws;
  bf16* hb   = (bf16*)(ws + 0);          // 16 MiB, LN1 out; reused as ctx
  bf16* ctxb = hb;
  bf16* qb   = (bf16*)(ws + 16 * MiB);   // 16 MiB
  bf16* kb   = (bf16*)(ws + 32 * MiB);   // 16 MiB
  bf16* vb   = (bf16*)(ws + 48 * MiB);   // 16 MiB
  float* x1  = (float*)(ws + 64 * MiB);  // 32 MiB post-attn residual
  bf16* h2b  = (bf16*)(ws + 96 * MiB);   // 16 MiB LN2 out
  bf16* ff1b = (bf16*)(ws + 112 * MiB);  // 64 MiB FFN hidden
  bf16* wqb  = (bf16*)(ws + 176 * MiB);  // 2 MiB
  bf16* wkb  = (bf16*)(ws + 178 * MiB);
  bf16* wvb  = (bf16*)(ws + 180 * MiB);
  bf16* wob  = (bf16*)(ws + 182 * MiB);
  bf16* w1b  = (bf16*)(ws + 184 * MiB);  // 8 MiB
  bf16* w2b  = (bf16*)(ws + 192 * MiB);  // 8 MiB

  // 1) weights fp32 -> bf16
  cvt_f32_bf16_kernel<<<2048, 256, 0, stream>>>(Wq, wqb, Dc * Dc);
  cvt_f32_bf16_kernel<<<2048, 256, 0, stream>>>(Wk, wkb, Dc * Dc);
  cvt_f32_bf16_kernel<<<2048, 256, 0, stream>>>(Wv, wvb, Dc * Dc);
  cvt_f32_bf16_kernel<<<2048, 256, 0, stream>>>(Wo, wob, Dc * Dc);
  cvt_f32_bf16_kernel<<<2048, 256, 0, stream>>>(W1, w1b, Dc * DFFc);
  cvt_f32_bf16_kernel<<<2048, 256, 0, stream>>>(W2, w2b, DFFc * Dc);

  // 2) LN1
  ln_bf16_kernel<<<Mc, 256, 0, stream>>>(x, ln1s, ln1b, hb);

  // 3) Q/K/V projections (8192x1024x1024), bf16 out
  dim3 gproj(Dc / 128, Mc / 128);
  gemm_wmma_kernel<false, false, false, false><<<gproj, 256, 0, stream>>>(
      hb, wqb, nullptr, nullptr, qb, Mc, Dc, Dc);
  gemm_wmma_kernel<false, false, false, false><<<gproj, 256, 0, stream>>>(
      hb, wkb, nullptr, nullptr, kb, Mc, Dc, Dc);
  gemm_wmma_kernel<false, false, false, false><<<gproj, 256, 0, stream>>>(
      hb, wvb, nullptr, nullptr, vb, Mc, Dc, Dc);

  // 4) causal flash attention -> ctx (reuses hb slot)
  flash_attn_kernel<<<dim3(Sc / 64, Hc, Bc), 128, 0, stream>>>(qb, kb, vb, ctxb);

  // 5) output projection + bias + residual -> x1 (fp32)
  gemm_wmma_kernel<true, false, true, true><<<gproj, 256, 0, stream>>>(
      ctxb, wob, bo, x, (void*)x1, Mc, Dc, Dc);

  // 6) LN2
  ln_bf16_kernel<<<Mc, 256, 0, stream>>>(x1, ln2s, ln2b, h2b);

  // 7) FFN1: 8192x4096x1024 + bias + gelu -> bf16
  dim3 gff1(DFFc / 128, Mc / 128);
  gemm_wmma_kernel<true, true, false, false><<<gff1, 256, 0, stream>>>(
      h2b, w1b, b1, nullptr, ff1b, Mc, DFFc, Dc);

  // 8) FFN2: 8192x1024x4096 + bias + residual -> d_out (fp32)
  gemm_wmma_kernel<true, false, true, true><<<gproj, 256, 0, stream>>>(
      ff1b, w2b, b2, x1, d_out, Mc, Dc, DFFc);
}
